// Mamba2_24283745091706
// MI455X (gfx1250) — matrix operand order repaired and checked by reading, not compiled, not run on hardware
//
#include <hip/hip_runtime.h>
#include <hip/hip_bf16.h>
#include <math.h>

// ---------------------------------------------------------------------------
// Mamba2 single-step decode for MI455X (gfx1250, wave32, WMMA).
//
// Roofline: ssm_state (537 MB) read dominates (~23us @ 23.3 TB/s). The two
// GEMMs (W_in 70MB, W_out 34MB) have AI ~128 FLOP/B -> need matrix-core rate
// -> v_wmma_f32_16x16x32_bf16 with split-bf16 (hi+lo) operands for ~f32
// accuracy: acc += ahi*bhi + alo*bhi + ahi*blo (3 WMMAs per 16x16x32 step).
// v2: 32x32 per-wave tile (12 WMMAs / K-step) to amortize the f32->bf16
// split conversions (VALU) over 4x the matrix work.
// ---------------------------------------------------------------------------

#define B_SZ     256
#define DMODEL   2048
#define DSTATE   128
#define DCONV    4
#define HEADDIM  64
#define DINNER   4096
#define NHEADS   64
#define CONVDIM  4352      // DINNER + 2*DSTATE
#define DINPROJ  8512      // 2*DINNER + 2*DSTATE + NHEADS
#define EPS_F    1e-5f

typedef __attribute__((ext_vector_type(16))) __bf16 v16bf;
typedef __attribute__((ext_vector_type(8)))  float  v8f;

#define WMMA_BF16(A, B, C) \
    __builtin_amdgcn_wmma_f32_16x16x32_bf16(false, (A), false, (B), (short)0, (C), false, false)

// Split a 16-float fragment into hi/lo bf16 halves (a = hi + lo, |lo| ~ 2^-8|a|).
__device__ __forceinline__ void cvt_split(const float f[16], v16bf& hi, v16bf& lo)
{
    #pragma unroll
    for (int i = 0; i < 16; ++i) {
        float  x = f[i];
        __bf16 h = (__bf16)x;
        hi[i] = h;
        lo[i] = (__bf16)(x - (float)h);
    }
}

// A 16x32 bf16 fragment (per CDNA5 ISA 7.12.2): lane holds row m = lane&15;
// element i<8 -> K = k0 + (lane>>4)*8 + i, element 8+i -> K = +16.
// `p` already includes row*K + (lane>>4)*8.
__device__ __forceinline__ void load_fragA(const float* __restrict__ p, int k0, float f[16])
{
    float4 a0 = *(const float4*)(p + k0);
    float4 a1 = *(const float4*)(p + k0 + 4);
    float4 a2 = *(const float4*)(p + k0 + 16);
    float4 a3 = *(const float4*)(p + k0 + 20);
    f[0]=a0.x;  f[1]=a0.y;  f[2]=a0.z;  f[3]=a0.w;
    f[4]=a1.x;  f[5]=a1.y;  f[6]=a1.z;  f[7]=a1.w;
    f[8]=a2.x;  f[9]=a2.y;  f[10]=a2.z; f[11]=a2.w;
    f[12]=a3.x; f[13]=a3.y; f[14]=a3.z; f[15]=a3.w;
}

// B 32x16 bf16 fragment: lane holds col n = lane&15; element i<8 -> K = k0 + (lane>>4)*8 + i, element 8+i -> K = +16.
// `p` already includes col*K + (lane>>4)*8.
__device__ __forceinline__ void load_fragB(const float* __restrict__ p, int k0, float f[16])
{
    float4 b0 = *(const float4*)(p + k0);
    float4 b1 = *(const float4*)(p + k0 + 4);
    float4 b2 = *(const float4*)(p + k0 + 16);
    float4 b3 = *(const float4*)(p + k0 + 20);
    f[0]=b0.x;  f[1]=b0.y;  f[2]=b0.z;  f[3]=b0.w;
    f[4]=b1.x;  f[5]=b1.y;  f[6]=b1.z;  f[7]=b1.w;
    f[8]=b2.x;  f[9]=b2.y;  f[10]=b2.z; f[11]=b2.w;
    f[12]=b3.x; f[13]=b3.y; f[14]=b3.z; f[15]=b3.w;
}

// ---------------------------------------------------------------------------
// GEMM (NT): C[M,N] = A[M,K] * B[N,K]^T, all f32 row-major in memory.
// Block: 8 waves as 4(M) x 2(N); wave tile 32x32 (4 accumulators, 12 WMMA/step).
// Requires M % 128 == 0, N % 64 == 0, K % 32 == 0 (true for all call sites).
// ---------------------------------------------------------------------------
__global__ __launch_bounds__(256) void gemm_nt_bf16x2(
    const float* __restrict__ A, const float* __restrict__ Bw,
    float* __restrict__ C, int M, int N, int K)
{
    const int lane  = threadIdx.x & 31;
    const int wave  = threadIdx.x >> 5;
    const int mBase = blockIdx.y * 128 + (wave >> 1) * 32;
    const int nBase = blockIdx.x * 64  + (wave & 1)  * 32;

    const int half = lane >> 4;             // 0: lanes 0-15, 1: lanes 16-31
    const int l16  = lane & 15;

    const float* arow0 = A  + (size_t)(mBase + l16)      * K + half * 8;
    const float* arow1 = A  + (size_t)(mBase + 16 + l16) * K + half * 8;
    const float* brow0 = Bw + (size_t)(nBase + l16)      * K + half * 8;
    const float* brow1 = Bw + (size_t)(nBase + 16 + l16) * K + half * 8;

    v8f acc00 = {}, acc01 = {}, acc10 = {}, acc11 = {};

    for (int k0 = 0; k0 < K; k0 += 32) {
        __builtin_prefetch(brow0 + k0 + 256, 0, 1);   // global_prefetch weight stream
        __builtin_prefetch(brow1 + k0 + 256, 0, 1);

        float fa0[16], fa1[16], fb[16];
        v16bf a0hi, a0lo, a1hi, a1lo, bhi, blo;

        load_fragA(arow0, k0, fa0);  cvt_split(fa0, a0hi, a0lo);
        load_fragA(arow1, k0, fa1);  cvt_split(fa1, a1hi, a1lo);

        // N-tile 0
        load_fragB(brow0, k0, fb);   cvt_split(fb, bhi, blo);
        acc00 = WMMA_BF16(a0hi, bhi, acc00);
        acc00 = WMMA_BF16(a0lo, bhi, acc00);
        acc00 = WMMA_BF16(a0hi, blo, acc00);
        acc10 = WMMA_BF16(a1hi, bhi, acc10);
        acc10 = WMMA_BF16(a1lo, bhi, acc10);
        acc10 = WMMA_BF16(a1hi, blo, acc10);

        // N-tile 1 (reuses A fragments; b0 staging regs dead by now)
        load_fragB(brow1, k0, fb);   cvt_split(fb, bhi, blo);
        acc01 = WMMA_BF16(a0hi, bhi, acc01);
        acc01 = WMMA_BF16(a0lo, bhi, acc01);
        acc01 = WMMA_BF16(a0hi, blo, acc01);
        acc11 = WMMA_BF16(a1hi, bhi, acc11);
        acc11 = WMMA_BF16(a1lo, bhi, acc11);
        acc11 = WMMA_BF16(a1hi, blo, acc11);
    }

    // C/D layout: lane -> col n = l16; VGPR j -> row m = j + 8*half.
    #pragma unroll
    for (int ti = 0; ti < 2; ++ti) {
        float* crow0 = C + (size_t)(mBase + ti * 16) * N + nBase + l16;
        float* crow1 = crow0 + 16;
        v8f acc_0 = ti ? acc10 : acc00;
        v8f acc_1 = ti ? acc11 : acc01;
        #pragma unroll
        for (int j = 0; j < 8; ++j) {
            crow0[(size_t)(j + half * 8) * N] = acc_0[j];
            crow1[(size_t)(j + half * 8) * N] = acc_1[j];
        }
    }
}

// ---------------------------------------------------------------------------
// Depthwise conv state step + bias + SiLU over CONVDIM channels.
// xBC[b,c] = silu( sum_i cs[b,c,i+1]*w[c,i] + xin[b,c]*w[c,3] + bias[c] )
// ---------------------------------------------------------------------------
__global__ __launch_bounds__(256) void conv_silu_kernel(
    const float* __restrict__ conv_state, const float* __restrict__ conv_w,
    const float* __restrict__ conv_b, const float* __restrict__ zxbcdt,
    float* __restrict__ xBC)
{
    int idx = blockIdx.x * 256 + threadIdx.x;
    if (idx >= B_SZ * CONVDIM) return;
    int b = idx / CONVDIM;
    int c = idx - b * CONVDIM;
    const float* cs = conv_state + (size_t)idx * DCONV;
    const float* w  = conv_w + (size_t)c * DCONV;
    float xin = zxbcdt[(size_t)b * DINPROJ + DINNER + c];
    float acc = cs[1] * w[0] + cs[2] * w[1] + cs[3] * w[2] + xin * w[3] + conv_b[c];
    acc = acc * (1.0f / (1.0f + expf(-acc)));          // SiLU
    xBC[idx] = acc;
}

// ---------------------------------------------------------------------------
// SSM: y[b,h,p] = dA * dot(ssm_state[b,h,p,:], C[b,:]) + x[p]*(dt*dot(B,C) + D[h])
// (fresh-input einsum term factorizes: sum_n dt*B[n]*x*C[n] = dt*x*(B.C))
// One block per (h,b); 64 threads = 2 waves; C,B staged in LDS; float4 streams.
// ---------------------------------------------------------------------------
__global__ __launch_bounds__(64) void ssm_kernel(
    const float* __restrict__ ssm_state, const float* __restrict__ zxbcdt,
    const float* __restrict__ xBC, const float* __restrict__ dt_bias,
    const float* __restrict__ A_log, const float* __restrict__ Dvec,
    float* __restrict__ Y)
{
    __shared__ float sB[DSTATE];
    __shared__ float sC[DSTATE];
    const int h = blockIdx.x;
    const int b = blockIdx.y;
    const int p = threadIdx.x;

    const float* xbc_b = xBC + (size_t)b * CONVDIM;
    sB[p]      = xbc_b[DINNER + p];
    sB[p + 64] = xbc_b[DINNER + 64 + p];
    sC[p]      = xbc_b[DINNER + DSTATE + p];
    sC[p + 64] = xbc_b[DINNER + DSTATE + 64 + p];
    __syncthreads();

    float dtr = zxbcdt[(size_t)b * DINPROJ + (DINNER + CONVDIM) + h] + dt_bias[h];
    float dt  = fmaxf(dtr, 0.0f) + log1pf(expf(-fabsf(dtr)));   // stable softplus
    float dA  = expf(dt * (-expf(A_log[h])));
    float xh  = xbc_b[h * HEADDIM + p];

    const float* ss = ssm_state + (((size_t)b * NHEADS + h) * HEADDIM + p) * DSTATE;
    float dot = 0.0f;
    #pragma unroll 8
    for (int n = 0; n < DSTATE; n += 4) {
        float4 s4 = *(const float4*)(ss + n);
        dot += s4.x * sC[n] + s4.y * sC[n + 1] + s4.z * sC[n + 2] + s4.w * sC[n + 3];
    }
    float sbc = 0.0f;
    #pragma unroll 8
    for (int n = 0; n < DSTATE; ++n) sbc += sB[n] * sC[n];

    Y[((size_t)b * NHEADS + h) * HEADDIM + p] = dA * dot + xh * (dt * sbc + Dvec[h]);
}

// ---------------------------------------------------------------------------
// Gated RMSNorm: y = y * silu(z); y *= rsqrt(mean(y^2)+eps) * norm_w  (in place)
// ---------------------------------------------------------------------------
__global__ __launch_bounds__(256) void gated_norm_kernel(
    const float* __restrict__ zxbcdt, const float* __restrict__ norm_w,
    float* __restrict__ Y)
{
    __shared__ float red[256];
    const int b = blockIdx.x;
    const int t = threadIdx.x;
    float*       yrow = Y + (size_t)b * DINNER;
    const float* zrow = zxbcdt + (size_t)b * DINPROJ;   // z = first DINNER cols

    float v[16];
    float ssum = 0.0f;
    #pragma unroll
    for (int j = 0; j < 16; ++j) {
        int idx = t + j * 256;
        float z  = zrow[idx];
        float g  = z * (1.0f / (1.0f + expf(-z)));
        float yv = yrow[idx] * g;
        v[j] = yv;
        ssum += yv * yv;
    }
    red[t] = ssum;
    __syncthreads();
    #pragma unroll
    for (int s = 128; s > 0; s >>= 1) {
        if (t < s) red[t] += red[t + s];
        __syncthreads();
    }
    float scale = rsqrtf(red[0] / (float)DINNER + EPS_F);
    #pragma unroll
    for (int j = 0; j < 16; ++j) {
        int idx = t + j * 256;
        yrow[idx] = v[j] * scale * norm_w[idx];
    }
}

// ---------------------------------------------------------------------------
extern "C" void kernel_launch(void* const* d_in, const int* in_sizes, int n_in,
                              void* d_out, int out_size, void* d_ws, size_t ws_size,
                              hipStream_t stream)
{
    const float* hs         = (const float*)d_in[0];   // (B,1,DMODEL)
    const float* conv_state = (const float*)d_in[1];   // (B,CONVDIM,DCONV)
    const float* ssm_state  = (const float*)d_in[2];   // (B,NHEADS,HEADDIM,DSTATE)
    const float* W_in       = (const float*)d_in[3];   // (DINPROJ,DMODEL)
    const float* conv_w     = (const float*)d_in[4];   // (CONVDIM,DCONV)
    const float* conv_b     = (const float*)d_in[5];   // (CONVDIM,)
    const float* dt_bias    = (const float*)d_in[6];   // (NHEADS,)
    const float* A_log      = (const float*)d_in[7];   // (NHEADS,)
    const float* Dv         = (const float*)d_in[8];   // (NHEADS,)
    const float* norm_w     = (const float*)d_in[9];   // (DINNER,)
    const float* W_out      = (const float*)d_in[10];  // (DMODEL,DINNER)
    float* out = (float*)d_out;

    float* zxbcdt = (float*)d_ws;                          // 256*8512
    float* xBC    = zxbcdt + (size_t)B_SZ * DINPROJ;       // 256*4352
    float* Yb     = xBC    + (size_t)B_SZ * CONVDIM;       // 256*4096

    // 1) in_proj: zxbcdt = h @ W_in^T   (M=256, N=8512, K=2048)
    gemm_nt_bf16x2<<<dim3(DINPROJ / 64, B_SZ / 128), 256, 0, stream>>>(
        hs, W_in, zxbcdt, B_SZ, DINPROJ, DMODEL);

    // 2) conv state step + SiLU
    conv_silu_kernel<<<dim3((B_SZ * CONVDIM + 255) / 256), 256, 0, stream>>>(
        conv_state, conv_w, conv_b, zxbcdt, xBC);

    // 3) SSM contraction (bandwidth-dominated: 537 MB of ssm_state)
    ssm_kernel<<<dim3(NHEADS, B_SZ), 64, 0, stream>>>(
        ssm_state, zxbcdt, xBC, dt_bias, A_log, Dv, Yb);

    // 4) gated RMSNorm (in place on Yb)
    gated_norm_kernel<<<dim3(B_SZ), 256, 0, stream>>>(zxbcdt, norm_w, Yb);

    // 5) out_proj: out = y @ W_out^T   (M=256, N=2048, K=4096)
    gemm_nt_bf16x2<<<dim3(DMODEL / 64, B_SZ / 128), 256, 0, stream>>>(
        Yb, W_out, out, B_SZ, DMODEL, DINNER);
}
